// LSTM_41317585388050
// MI455X (gfx1250) — matrix operand order repaired and checked by reading, not compiled, not run on hardware
//
#include <hip/hip_runtime.h>
#include <stdint.h>

// LSTM scan, persistent-kernel formulation for MI455X (gfx1250, wave32, WMMA).
// 16 workgroups x 256 threads (8 wave32s). WG `w` owns hidden columns
// [16w, 16w+16) and the corresponding 64 gate columns; per step it runs a
// M=128 x N=64 x K=(256+64) bf16 WMMA GEMM, applies gates, updates its c/h
// slice, and exchanges h through a double-buffered global bf16 buffer with an
// agent-scope atomic grid barrier. The readout GEMM (h @ W_out^T + b_out) for
// step t-1 is fused into step t on waves 0-1 (reads the same h buffer the
// recurrent GEMM needs anyway). x rows for step t+1 are prefetched during
// step t (global_prefetch_b8) to hide the only HBM-cold traffic.

constexpr int kSeq     = 2048;
constexpr int kB       = 128;
constexpr int kI       = 64;
constexpr int kH       = 256;
constexpr int kO       = 64;
constexpr int kNwg     = 16;
constexpr int kThreads = 256;   // 8 wave32s

typedef __attribute__((ext_vector_type(16))) __bf16 v16bf;
typedef __attribute__((ext_vector_type(2)))  __bf16 v2bf;
typedef __attribute__((ext_vector_type(8)))  float  v8f;

union ABu { uint32_t u[8]; v16bf v; };

__device__ __forceinline__ uint32_t f2bf(float f) {
  uint32_t u = __builtin_bit_cast(uint32_t, f);
  return (u + 0x7FFFu + ((u >> 16) & 1u)) >> 16;  // round-to-nearest-even
}
__device__ __forceinline__ uint32_t pack2bf(float lo, float hi) {
#if __has_builtin(__builtin_amdgcn_cvt_pk_bf16_f32)
  v2bf p = __builtin_amdgcn_cvt_pk_bf16_f32(lo, hi);   // v_cvt_pk_bf16_f32
  return __builtin_bit_cast(uint32_t, p);
#else
  return f2bf(lo) | (f2bf(hi) << 16);
#endif
}
__device__ __forceinline__ float fsigmoid(float x) {
  return 1.0f / (1.0f + __expf(-x));
}
__device__ __forceinline__ float ftanh(float x) {
  float e = __expf(2.0f * fabsf(x));             // inf-safe: 2/(inf+1) -> 0
  return copysignf(1.0f - 2.0f / (e + 1.0f), x);
}

// Device-wide barrier: release own stores to agent scope, one lane per WG
// bumps the counter, everyone re-acquires (invalidates stale WGP$).
__device__ __forceinline__ void grid_sync(unsigned* ctr, unsigned target) {
  __builtin_amdgcn_fence(__ATOMIC_RELEASE, "agent");
  __syncthreads();
  if (threadIdx.x == 0) {
    __hip_atomic_fetch_add(ctr, 1u, __ATOMIC_RELAXED, __HIP_MEMORY_SCOPE_AGENT);
    while (__hip_atomic_load(ctr, __ATOMIC_RELAXED, __HIP_MEMORY_SCOPE_AGENT) < target)
      __builtin_amdgcn_s_sleep(2);
  }
  __syncthreads();
  __builtin_amdgcn_fence(__ATOMIC_ACQUIRE, "agent");
}

// One 16x16 tile of out[t] = h_t @ W_out^T + b_out. `tile` in [0,32):
// mt = tile/4 (batch tile), nt = tile%4 (16 output cols).
__device__ __forceinline__ void readout_tile(const uint32_t* hp32,
                                             const uint32_t* sWout,
                                             const float* sBout,
                                             float* op, int tile,
                                             int n16, int khi) {
  const int mt = tile >> 2, nt = tile & 3;
  v8f acc;
  const float bb = sBout[nt * 16 + n16];
#pragma unroll
  for (int r = 0; r < 8; ++r) acc[r] = bb;
  const uint32_t* hrow = hp32 + (mt * 16 + n16) * (kH / 2);
#pragma unroll 2
  for (int kb = 0; kb < 8; ++kb) {
    ABu a, b;
#pragma unroll
    for (int v = 0; v < 8; ++v)   // A 16x32 bf16 layout (ISA 7.12.2)
      a.u[v] = hrow[kb * 16 + (v >> 2) * 8 + khi * 4 + (v & 3)];
#pragma unroll
    for (int v = 0; v < 8; ++v)   // B 32x16 bf16 layout, k-pair packed
      b.u[v] = sWout[(kb * 16 + (v >> 2) * 8 + khi * 4 + (v & 3)) * 64 + nt * 16 + n16];
    acc = __builtin_amdgcn_wmma_f32_16x16x32_bf16(false, a.v, false, b.v,
                                                  (short)0, acc, false, false);
  }
#pragma unroll
  for (int r = 0; r < 8; ++r)
    op[(mt * 16 + r + 8 * khi) * kO + nt * 16 + n16] = acc[r];
}

__global__ __launch_bounds__(kThreads, 1)
void lstm_persistent(const float* __restrict__ x,
                     const float* __restrict__ h0,
                     const float* __restrict__ c0,
                     const float* __restrict__ W_ih,
                     const float* __restrict__ W_hh,
                     const float* __restrict__ bvec,
                     const float* __restrict__ W_out,
                     const float* __restrict__ b_out,
                     float* __restrict__ out,
                     unsigned* __restrict__ barrier_ctr,
                     uint16_t* __restrict__ hbase) {
  // bf16 weight slices, pre-packed as (k, k+1) dword pairs per column so each
  // lane's B fragment is 8 dword LDS loads. ~74 KB of the 320 KB WGP LDS.
  __shared__ uint32_t sWhh[(kH / 2) * 64];   // 32 KB: W_hh^T slice (K=256, N=64)
  __shared__ uint32_t sWih[(kI / 2) * 64];   //  8 KB: W_ih^T slice (K=64,  N=64)
  __shared__ uint32_t sWout[(kH / 2) * 64];  // 32 KB: W_out^T     (K=256, N=64)
  __shared__ float    sBias[64];
  __shared__ float    sBout[64];

  const int tid  = threadIdx.x;
  const int wg   = blockIdx.x;      // 0..15
  const int j0   = wg * 16;         // this WG's hidden-column base
  const int lane = tid & 31;
  const int wave = tid >> 5;        // 0..7 == batch tile (16 rows each)
  const int n16  = lane & 15;       // tile column / A row
  const int khi  = lane >> 4;       // lane-half selector in WMMA layouts

  // ---- one-time setup: pack weights into LDS -------------------------------
  for (int idx = tid; idx < (kH / 2) * 64; idx += kThreads) {
    const int kk = idx >> 6, n = idx & 63;
    const int gc = (n >> 4) * kH + j0 + (n & 15);   // gate row (i,f,g,o blocks)
    sWhh[idx] = pack2bf(W_hh[gc * kH + 2 * kk], W_hh[gc * kH + 2 * kk + 1]);
  }
  for (int idx = tid; idx < (kI / 2) * 64; idx += kThreads) {
    const int kk = idx >> 6, n = idx & 63;
    const int gc = (n >> 4) * kH + j0 + (n & 15);
    sWih[idx] = pack2bf(W_ih[gc * kI + 2 * kk], W_ih[gc * kI + 2 * kk + 1]);
  }
  for (int idx = tid; idx < (kH / 2) * 64; idx += kThreads) {
    const int kk = idx >> 6, o = idx & 63;
    sWout[idx] = pack2bf(W_out[o * kH + 2 * kk], W_out[o * kH + 2 * kk + 1]);
  }
  if (tid < 64) {
    const int gc = (tid >> 4) * kH + j0 + (tid & 15);
    sBias[tid] = bvec[gc];
    sBout[tid] = b_out[tid];
  }
  __syncthreads();

  uint16_t* hb0 = hbase;            // h ping-pong (bf16, 128x256 each)
  uint16_t* hb1 = hbase + kB * kH;

  // seed h(0) (this WG's 16 columns) and c state (WMMA-C register layout)
  for (int idx = tid; idx < kB * 16; idx += kThreads) {
    const int b = idx >> 4, col = j0 + (idx & 15);
    hb0[b * kH + col] = (uint16_t)f2bf(h0[b * kH + col]);
  }
  v8f c;
#pragma unroll
  for (int r = 0; r < 8; ++r)
    c[r] = c0[(wave * 16 + r + 8 * khi) * kH + j0 + n16];

  grid_sync(barrier_ctr, kNwg);     // h(0) visible everywhere

  // this lane's x row base (row = wave*16 + n16, 64 floats = 256 B)
  const float* xlane = x + (size_t)(wave * 16 + n16) * kI;

  // ---- sequential scan -----------------------------------------------------
  for (int t = 0; t < kSeq; ++t) {
    const uint16_t* hprev = (t & 1) ? hb1 : hb0;
    uint16_t*       hnext = (t & 1) ? hb0 : hb1;
    const uint32_t* hp32  = (const uint32_t*)hprev;

    // prefetch next step's x row while this step's WMMAs run
    if (t + 1 < kSeq) {
      const char* nx = (const char*)(xlane + (size_t)(t + 1) * (kB * kI));
      __builtin_prefetch(nx, 0, 3);        // global_prefetch_b8
      __builtin_prefetch(nx + 128, 0, 3);
    }

    // fused readout of out[t-1] (waves 0-1; h_{t-1} is hot in cache anyway)
    if (t > 0 && wave < 2)
      readout_tile(hp32, sWout, sBout, out + (size_t)(t - 1) * (kB * kO),
                   wg * 2 + wave, n16, khi);

    // gate accumulators, seeded with bias (bias depends only on column)
    v8f acc[4];
#pragma unroll
    for (int g = 0; g < 4; ++g) {
      const float bg = sBias[g * 16 + n16];
#pragma unroll
      for (int r = 0; r < 8; ++r) acc[g][r] = bg;
    }

    // input projection: x_t(16x64 rows of this wave) @ W_ih^T, K=64
    const float2* xrow = (const float2*)(xlane + (size_t)t * (kB * kI));
#pragma unroll
    for (int kb = 0; kb < 2; ++kb) {
      ABu a;
#pragma unroll
      for (int v = 0; v < 8; ++v) {
        const int kd = (kb * 32 + (v >> 2) * 16 + khi * 8 + (v & 3) * 2) >> 1;
        const float2 f = xrow[kd];
        a.u[v] = pack2bf(f.x, f.y);
      }
#pragma unroll
      for (int g = 0; g < 4; ++g) {
        ABu b;
#pragma unroll
        for (int v = 0; v < 8; ++v)
          b.u[v] = sWih[(kb * 16 + (v >> 2) * 8 + khi * 4 + (v & 3)) * 64 + g * 16 + n16];
        acc[g] = __builtin_amdgcn_wmma_f32_16x16x32_bf16(
            false, a.v, false, b.v, (short)0, acc[g], false, false);
      }
    }

    // recurrence: h_{t-1} @ W_hh^T, K=256 (A streamed from L2, B from LDS)
    const uint32_t* hrow = hp32 + (wave * 16 + n16) * (kH / 2);
#pragma unroll 2
    for (int kb = 0; kb < 8; ++kb) {
      ABu a;
#pragma unroll
      for (int v = 0; v < 8; ++v)
        a.u[v] = hrow[kb * 16 + (v >> 2) * 8 + khi * 4 + (v & 3)];
#pragma unroll
      for (int g = 0; g < 4; ++g) {
        ABu b;
#pragma unroll
        for (int v = 0; v < 8; ++v)
          b.u[v] = sWhh[(kb * 16 + (v >> 2) * 8 + khi * 4 + (v & 3)) * 64 + g * 16 + n16];
        acc[g] = __builtin_amdgcn_wmma_f32_16x16x32_bf16(
            false, a.v, false, b.v, (short)0, acc[g], false, false);
      }
    }

    // gates + state update (f32), publish h_t slice as bf16
#pragma unroll
    for (int r = 0; r < 8; ++r) {
      const float ig = fsigmoid(acc[0][r]);
      const float fg = fsigmoid(acc[1][r]);
      const float gg = ftanh(acc[2][r]);
      const float og = fsigmoid(acc[3][r]);
      const float cn = fg * c[r] + ig * gg;
      c[r] = cn;
      const float hv = og * ftanh(cn);
      hnext[(wave * 16 + r + 8 * khi) * kH + j0 + n16] =
          (uint16_t)(pack2bf(hv, 0.0f) & 0xFFFFu);
    }

    grid_sync(barrier_ctr, (unsigned)(t + 2) * kNwg);
  }

  // readout for the final step (h_{SEQ-1} lives in hb0 since kSeq is even)
  if (wave < 2)
    readout_tile((const uint32_t*)hb0, sWout, sBout,
                 out + (size_t)(kSeq - 1) * (kB * kO), wg * 2 + wave, n16, khi);
}

extern "C" void kernel_launch(void* const* d_in, const int* in_sizes, int n_in,
                              void* d_out, int out_size, void* d_ws,
                              size_t ws_size, hipStream_t stream) {
  (void)in_sizes; (void)n_in; (void)out_size; (void)ws_size;
  const float* x     = (const float*)d_in[0];
  const float* h0    = (const float*)d_in[1];
  const float* c0    = (const float*)d_in[2];
  const float* W_ih  = (const float*)d_in[3];
  const float* W_hh  = (const float*)d_in[4];
  const float* bvec  = (const float*)d_in[5];
  const float* W_out = (const float*)d_in[6];
  const float* b_out = (const float*)d_in[7];
  float* out = (float*)d_out;

  unsigned* ctr   = (unsigned*)d_ws;                      // barrier counter
  uint16_t* hbase = (uint16_t*)((char*)d_ws + 256);       // 2 x 64 KB h ping-pong

  hipMemsetAsync(d_ws, 0, 256, stream);                   // zero counter only

  lstm_persistent<<<dim3(kNwg), dim3(kThreads), 0, stream>>>(
      x, h0, c0, W_ih, W_hh, bvec, W_out, b_out, out, ctr, hbase);
}
